// Attention_21552145891894
// MI455X (gfx1250) — compile-verified, matrix operand order checked by reading, not run on hardware
//
#include <hip/hip_runtime.h>
#include <stdint.h>

#define B_    64
#define S_    1024
#define DEC_  1024
#define ENC_  1024
#define ATTN_ 1024

typedef __attribute__((ext_vector_type(16))) __bf16 v16bf;
typedef __attribute__((ext_vector_type(8)))  float  v8f;

__device__ __forceinline__ uint32_t f32_to_bf16_bits(float f) {
    uint32_t x = __float_as_uint(f);
    return (x + 0x7FFFu + ((x >> 16) & 1u)) >> 16;   // round-to-nearest-even
}
__device__ __forceinline__ uint32_t pack_bf16(float lo, float hi) {
    return f32_to_bf16_bits(lo) | (f32_to_bf16_bits(hi) << 16);
}

// ---------------------------------------------------------------------------
// Kernel 1: W_e = W_attn[:, DEC:DEC+ENC] -> bf16, row-major [ATTN][ENC]
// grid(ATTN), block(256): each thread converts 4 elements of one row.
// ---------------------------------------------------------------------------
__global__ __launch_bounds__(256) void cvt_we_kernel(const float* __restrict__ W,
                                                     uint32_t* __restrict__ wsW) {
    const int a = blockIdx.x;
    const int t = threadIdx.x;
    const float4* src = (const float4*)(W + (size_t)a * (DEC_ + ENC_) + DEC_);
    float4 f = src[t];
    uint2 p;
    p.x = pack_bf16(f.x, f.y);
    p.y = pack_bf16(f.z, f.w);
    ((uint2*)(wsW + (size_t)a * (ENC_ / 2)))[t] = p;
}

// ---------------------------------------------------------------------------
// Kernel 2: hW[b][a] = dot(W_attn[a, 0:DEC], hidden[b]) + b_attn[a]
// grid(ATTN/256, B), block(256). hidden row cached in LDS.
// ---------------------------------------------------------------------------
__global__ __launch_bounds__(256) void hproj_kernel(const float* __restrict__ W,
                                                    const float* __restrict__ hidden,
                                                    const float* __restrict__ b_attn,
                                                    float* __restrict__ hW) {
    __shared__ float hl[DEC_];
    const int b = blockIdx.y;
    const int t = threadIdx.x;
    ((float4*)hl)[t] = ((const float4*)(hidden + (size_t)b * DEC_))[t];
    __syncthreads();
    const int a = blockIdx.x * 256 + t;
    const float4* wr = (const float4*)(W + (size_t)a * (DEC_ + ENC_));
    const float4* h4 = (const float4*)hl;
    float acc = 0.f;
    for (int i = 0; i < DEC_ / 4; ++i) {
        float4 w4 = wr[i];
        float4 hh = h4[i];
        acc += w4.x * hh.x + w4.y * hh.y + w4.z * hh.z + w4.w * hh.w;
    }
    hW[(size_t)b * ATTN_ + a] = acc + b_attn[a];
}

// ---------------------------------------------------------------------------
// Kernel 3: fused scores GEMM, K-outer loop.
// scores[b][s] = sum_a v[a] * tanh( enc[b,s,:]·W_e[a,:] + hW[b][a] )
// grid(S/32, B), block(256) = 8 waves. A tile (32 x 1024 bf16) in LDS.
// Wave w owns N columns [w*128, w*128+128): 8 sub-tiles of 16, accumulators
// c0[8]/c1[8] (128 VGPRs) held across the K loop. Per K-step: 2 A fragments
// from LDS feed 16 wmma against 8 streamed B fragments -> no loop-invariant
// A register blob, no scratch spills. Epilogue after the K loop: tanh(c+hW)*v
// with deterministic ds_swizzle-xor butterfly + fixed-order 8-way LDS combine.
// ---------------------------------------------------------------------------
__global__ __launch_bounds__(256) void scores_kernel(const float* __restrict__ enc,
                                                     const uint32_t* __restrict__ wsW,
                                                     const float* __restrict__ hW,
                                                     const float* __restrict__ vvec,
                                                     float* __restrict__ scores) {
    __shared__ uint32_t AldsU[32 * (ENC_ / 2)];   // 32 rows x 1024 bf16 = 64 KB
    __shared__ float    sbuf[8][32];

    const int b   = blockIdx.y;
    const int s0  = blockIdx.x * 32;
    const int tid = threadIdx.x;

    // Stage enc tile -> bf16 LDS (32*256 float4 total, 32 per thread).
    const float4* esrc = (const float4*)(enc + ((size_t)b * S_ + s0) * ENC_);
    for (int it = 0; it < 32; ++it) {
        const int p = tid + it * 256;
        float4 f = esrc[p];
        uint2 q;
        q.x = pack_bf16(f.x, f.y);
        q.y = pack_bf16(f.z, f.w);
        ((uint2*)AldsU)[p] = q;
    }
    __syncthreads();

    const int wave = tid >> 5;          // N-strip: columns [wave*128, wave*128+128)
    const int lane = tid & 31;
    const int half = lane >> 4;         // selects K-run (+0 / +8) per ISA A/B layout
    const int l16  = lane & 15;

    const uint32_t* arow0 = AldsU + (l16)      * (ENC_ / 2);   // M-tile 0 rows
    const uint32_t* arow1 = AldsU + (16 + l16) * (ENC_ / 2);   // M-tile 1 rows
    const uint32_t* bbase = wsW + (size_t)(wave * 128 + l16) * (ENC_ / 2);
    const float*    hWrow = hW + (size_t)b * ATTN_;

    v8f c0[8], c1[8];
#pragma unroll
    for (int j = 0; j < 8; ++j) {
        c0[j] = (v8f){0.f, 0.f, 0.f, 0.f, 0.f, 0.f, 0.f, 0.f};
        c1[j] = (v8f){0.f, 0.f, 0.f, 0.f, 0.f, 0.f, 0.f, 0.f};
    }

#pragma unroll 1
    for (int k = 0; k < ENC_; k += 32) {
        const int u0 = (k + half * 8) >> 1;   // uint offset, 16B aligned
        union { uint4 q[2]; v16bf v; } fa0, fa1;
        fa0.q[0] = *(const uint4*)(arow0 + u0);
        fa0.q[1] = *(const uint4*)(arow0 + u0 + 8);    // K += 16
        fa1.q[0] = *(const uint4*)(arow1 + u0);
        fa1.q[1] = *(const uint4*)(arow1 + u0 + 8);
#pragma unroll
        for (int j = 0; j < 8; ++j) {
            union { uint4 q[2]; v16bf v; } fb;
            const uint32_t* brow = bbase + j * 16 * (ENC_ / 2);
            fb.q[0] = *(const uint4*)(brow + u0);
            fb.q[1] = *(const uint4*)(brow + u0 + 8);
            c0[j] = __builtin_amdgcn_wmma_f32_16x16x32_bf16(false, fa0.v, false, fb.v,
                                                            (short)0, c0[j], false, false);
            c1[j] = __builtin_amdgcn_wmma_f32_16x16x32_bf16(false, fa1.v, false, fb.v,
                                                            (short)0, c1[j], false, false);
        }
    }

    float rs0[8], rs1[8];
#pragma unroll
    for (int r = 0; r < 8; ++r) { rs0[r] = 0.f; rs1[r] = 0.f; }

#pragma unroll
    for (int j = 0; j < 8; ++j) {
        const int   nn = wave * 128 + j * 16 + l16;   // C layout: N = lane&15
        const float hv = hWrow[nn];
        const float vv = vvec[nn];
#pragma unroll
        for (int r = 0; r < 8; ++r) {   // C layout: M = r + 8*half
            float t0 = tanhf(c0[j][r] + hv) * vv;
            t0 += __int_as_float(__builtin_amdgcn_ds_swizzle(__float_as_int(t0), (1 << 10) | 0x1f));
            t0 += __int_as_float(__builtin_amdgcn_ds_swizzle(__float_as_int(t0), (2 << 10) | 0x1f));
            t0 += __int_as_float(__builtin_amdgcn_ds_swizzle(__float_as_int(t0), (4 << 10) | 0x1f));
            t0 += __int_as_float(__builtin_amdgcn_ds_swizzle(__float_as_int(t0), (8 << 10) | 0x1f));
            rs0[r] += t0;               // valid in lanes 0 and 16

            float t1 = tanhf(c1[j][r] + hv) * vv;
            t1 += __int_as_float(__builtin_amdgcn_ds_swizzle(__float_as_int(t1), (1 << 10) | 0x1f));
            t1 += __int_as_float(__builtin_amdgcn_ds_swizzle(__float_as_int(t1), (2 << 10) | 0x1f));
            t1 += __int_as_float(__builtin_amdgcn_ds_swizzle(__float_as_int(t1), (4 << 10) | 0x1f));
            t1 += __int_as_float(__builtin_amdgcn_ds_swizzle(__float_as_int(t1), (8 << 10) | 0x1f));
            rs1[r] += t1;
        }
    }

    if (l16 == 0) {                     // lane 0: rows +0..7, lane 16: rows +8..15
#pragma unroll
        for (int r = 0; r < 8; ++r) {
            sbuf[wave][half * 8 + r]      = rs0[r];   // rows 0..15
            sbuf[wave][16 + half * 8 + r] = rs1[r];   // rows 16..31
        }
    }
    __syncthreads();

    if (tid < 32) {                     // fixed-order 8-way combine per row
        const int m = tid;
        float s = 0.f;
#pragma unroll
        for (int w = 0; w < 8; ++w) s += sbuf[w][m];
        scores[(size_t)b * S_ + s0 + m] = s;
    }
}

// ---------------------------------------------------------------------------
// Kernel 4: softmax over S + context = sum_s w[s] * enc[b,s,:]
// grid(B), block(256). Fixed-order tree reductions -> deterministic.
// ---------------------------------------------------------------------------
__global__ __launch_bounds__(256) void ctx_kernel(const float* __restrict__ scores,
                                                  const float* __restrict__ enc,
                                                  float* __restrict__ out) {
    __shared__ float w[S_];
    __shared__ float red[256];
    const int b = blockIdx.x;
    const int t = threadIdx.x;

    float4 sv = ((const float4*)(scores + (size_t)b * S_))[t];
    red[t] = fmaxf(fmaxf(sv.x, sv.y), fmaxf(sv.z, sv.w));
    __syncthreads();
    for (int off = 128; off > 0; off >>= 1) {
        if (t < off) red[t] = fmaxf(red[t], red[t + off]);
        __syncthreads();
    }
    const float mx = red[0];
    __syncthreads();

    sv.x = __expf(sv.x - mx); sv.y = __expf(sv.y - mx);
    sv.z = __expf(sv.z - mx); sv.w = __expf(sv.w - mx);
    red[t] = sv.x + sv.y + sv.z + sv.w;
    __syncthreads();
    for (int off = 128; off > 0; off >>= 1) {
        if (t < off) red[t] += red[t + off];
        __syncthreads();
    }
    const float inv = 1.0f / red[0];
    ((float4*)w)[t] = make_float4(sv.x * inv, sv.y * inv, sv.z * inv, sv.w * inv);
    __syncthreads();

    float4 acc = make_float4(0.f, 0.f, 0.f, 0.f);
    const float4* e4 = (const float4*)(enc + (size_t)b * S_ * ENC_);
    for (int s = 0; s < S_; ++s) {
        const float ws = w[s];
        float4 ev = e4[(size_t)s * (ENC_ / 4) + t];
        acc.x += ws * ev.x; acc.y += ws * ev.y;
        acc.z += ws * ev.z; acc.w += ws * ev.w;
    }
    ((float4*)(out + (size_t)b * ENC_))[t] = acc;
}

// ---------------------------------------------------------------------------
extern "C" void kernel_launch(void* const* d_in, const int* in_sizes, int n_in,
                              void* d_out, int out_size, void* d_ws, size_t ws_size,
                              hipStream_t stream) {
    (void)in_sizes; (void)n_in; (void)out_size; (void)ws_size;
    const float* hidden = (const float*)d_in[0];   // [B, DEC]
    const float* enc    = (const float*)d_in[1];   // [B, S, ENC]
    const float* W      = (const float*)d_in[2];   // [ATTN, DEC+ENC]
    const float* b_attn = (const float*)d_in[3];   // [ATTN]
    const float* vvec   = (const float*)d_in[4];   // [ATTN]
    float* out = (float*)d_out;                    // [B, 1, ENC]

    uint8_t* ws = (uint8_t*)d_ws;
    uint32_t* wsW    = (uint32_t*)ws;                                   // 2 MB bf16 W_e
    float*    hW     = (float*)(ws + (size_t)ATTN_ * ENC_ * 2);         // 256 KB
    float*    scores = (float*)(ws + (size_t)ATTN_ * ENC_ * 2
                                   + (size_t)B_ * ATTN_ * 4);           // 256 KB

    hipLaunchKernelGGL(cvt_we_kernel, dim3(ATTN_), dim3(256), 0, stream, W, wsW);
    hipLaunchKernelGGL(hproj_kernel, dim3(ATTN_ / 256, B_), dim3(256), 0, stream,
                       W, hidden, b_attn, hW);
    hipLaunchKernelGGL(scores_kernel, dim3(S_ / 32, B_), dim3(256), 0, stream,
                       enc, wsW, hW, vvec, scores);
    hipLaunchKernelGGL(ctx_kernel, dim3(B_), dim3(256), 0, stream, scores, enc, out);
}
